// EvoAttn_2190433321003
// MI455X (gfx1250) — matrix operand order repaired and checked by reading, not compiled, not run on hardware
//
#include <hip/hip_runtime.h>

#define B_ 2
#define L_ 2048
#define E_ 1024
#define H_ 16
#define D_ 64

typedef __attribute__((ext_vector_type(16))) unsigned short v16u;
typedef __attribute__((ext_vector_type(16))) __bf16 v16bf;
typedef __attribute__((ext_vector_type(8)))  float  v8f;
typedef __attribute__((ext_vector_type(4)))  unsigned int u32x4;
typedef __attribute__((ext_vector_type(8)))  int i32x8;
typedef __attribute__((ext_vector_type(4)))  int i32x4;

union Frag {
  v16u v;
  uint4 q[2];
  unsigned short s[16];
};

// float -> bf16 bits, round-to-nearest-even
__device__ __forceinline__ unsigned short f2bf(float f) {
  unsigned u = __builtin_bit_cast(unsigned, f);
  return (unsigned short)((u + 0x7FFFu + ((u >> 16) & 1u)) >> 16);
}

__device__ __forceinline__ v8f wmma_bf16(v16u a, v16u b, v8f c) {
  return __builtin_amdgcn_wmma_f32_16x16x32_bf16(
      false, __builtin_bit_cast(v16bf, a),
      false, __builtin_bit_cast(v16bf, b),
      (short)0, c, false, false);
}

// ---------------------------------------------------------------------------
// TDM: async-load a 64-row x 32-elem (bf16) tile, row stride 1024 elems, from
// global to LDS. D# built per CDNA5 ISA 8.3/8.4 (2D tensor, no pad, no gather).
// Wave-level instruction: call from one wave only; tracked by TENSORcnt.
// ---------------------------------------------------------------------------
__device__ __forceinline__ void tdm_load_w_tile(void* lds_dst,
                                                const unsigned short* gsrc) {
  const unsigned long long ga = (unsigned long long)gsrc;
  u32x4 g0;
  g0[0] = 1u;                                   // count=1, user mode
  g0[1] = (unsigned)(unsigned long long)lds_dst;  // LDS byte address
  g0[2] = (unsigned)ga;                         // global_addr[31:0]
  g0[3] = (unsigned)(ga >> 32) | 0x80000000u;   // global_addr[56:32] | type=2
  i32x8 g1;
  g1[0] = 0x00010000;            // wg_mask=0, data_size=1 (2 bytes), flags=0
  g1[1] = (int)(1024u << 16);    // tensor_dim0[15:0]=1024 (atomic addr = 0)
  g1[2] = (int)0xFFFF0000u;      // tensor_dim0[31:16]=0, tensor_dim1[15:0]=65535
  g1[3] = (int)(32u << 16);      // tensor_dim1[31:16]=0, tile_dim0=32
  g1[4] = 64;                    // tile_dim1=64, tile_dim2=0
  g1[5] = 1024;                  // tensor_dim0_stride[31:0]=1024
  g1[6] = 0;                     // stride hi / dim1_stride lo
  g1[7] = 0;
  i32x4 g2 = {0, 0, 0, 0};
  i32x4 g3 = {0, 0, 0, 0};
#if defined(__clang_major__) && (__clang_major__ >= 23)
  i32x8 g4 = {0, 0, 0, 0, 0, 0, 0, 0};
  __builtin_amdgcn_tensor_load_to_lds(g0, g1, g2, g3, g4, 0);
#else
  __builtin_amdgcn_tensor_load_to_lds(g0, g1, g2, g3, 0);
#endif
}

// ---------------------------------------------------------------------------
// Kernel 0a: convert the four E x E weight matrices to bf16 (one-time).
// ---------------------------------------------------------------------------
__global__ __launch_bounds__(256) void cvt_w_kernel(
    const float* __restrict__ Wq, const float* __restrict__ Wk,
    const float* __restrict__ Wv, const float* __restrict__ Wo,
    unsigned short* __restrict__ out) {
  const int z = blockIdx.y;
  const float* W = (z == 0) ? Wq : (z == 1) ? Wk : (z == 2) ? Wv : Wo;
  unsigned short* o = out + (size_t)z * (E_ * E_);
  const size_t i = ((size_t)blockIdx.x * 256 + threadIdx.x) * 8;
  const float4* p = (const float4*)(W + i);
  float4 a = p[0], b = p[1];
  Frag t;
  t.s[0] = f2bf(a.x); t.s[1] = f2bf(a.y); t.s[2] = f2bf(a.z); t.s[3] = f2bf(a.w);
  t.s[4] = f2bf(b.x); t.s[5] = f2bf(b.y); t.s[6] = f2bf(b.z); t.s[7] = f2bf(b.w);
  *(uint4*)(o + i) = t.q[0];
}

// ---------------------------------------------------------------------------
// Kernel 0b: xb = bf16(x * mask_row)  (mask folded in once; (x*m)@W^T==(x@W^T)*m)
// ---------------------------------------------------------------------------
__global__ __launch_bounds__(256) void cvt_x_kernel(
    const float* __restrict__ x, const float* __restrict__ amask,
    unsigned short* __restrict__ xb) {
  const size_t i = ((size_t)blockIdx.x * 256 + threadIdx.x) * 8;
  const float am = amask[i >> 10];  // row = i / E_
  const float4* p = (const float4*)(x + i);
  float4 a = p[0], b = p[1];
  Frag t;
  t.s[0] = f2bf(a.x * am); t.s[1] = f2bf(a.y * am);
  t.s[2] = f2bf(a.z * am); t.s[3] = f2bf(a.w * am);
  t.s[4] = f2bf(b.x * am); t.s[5] = f2bf(b.y * am);
  t.s[6] = f2bf(b.z * am); t.s[7] = f2bf(b.w * am);
  *(uint4*)(xb + i) = t.q[0];
}

// ---------------------------------------------------------------------------
// Kernel 1: fused QKV projection, all-bf16 operands, TDM-staged weights.
//   z==0 -> Q (scaled by 1/sqrt(D)), layout (B,H,L,D)
//   z==1 -> K, layout (B,H,L,D)
//   z==2 -> V, layout (B,H,D,L)  (transposed for contiguous PV B-frags)
// Block: 256 threads (8 waves), tile 128(M) x 64(N), K-steps of 32,
// double-buffered TENSOR_LOAD_TO_LDS for the 64x32 weight tile.
// ---------------------------------------------------------------------------
__global__ __launch_bounds__(256) void qkv_proj_kernel(
    const unsigned short* __restrict__ xb,
    const unsigned short* __restrict__ Wqb,
    const unsigned short* __restrict__ Wkb,
    const unsigned short* __restrict__ Wvb,
    unsigned short* __restrict__ Qb, unsigned short* __restrict__ Kb,
    unsigned short* __restrict__ Vb) {
  __shared__ __align__(16) unsigned short ldsW[2][64 * 32];
  const int z = blockIdx.z;
  const unsigned short* Wb = (z == 0) ? Wqb : ((z == 1) ? Wkb : Wvb);
  const float oscale = (z == 0) ? 0.125f : 1.0f;  // 1/sqrt(64)
  const int m0 = blockIdx.x * 128;
  const int n0 = blockIdx.y * 64;
  const int tid = threadIdx.x;
  const int wave = tid >> 5;
  const int lane = tid & 31;
  const int lhalf = lane >> 4;
  const int l15 = lane & 15;
  const int aRow = m0 + wave * 16 + l15;  // global row (b*L + l)
  const unsigned short* Wtile = Wb + (size_t)n0 * E_;

  v8f acc[4] = {};
  const int NIT = E_ / 32;

  for (int it = 0; it < NIT; ++it) {
    const int k0 = it * 32;
    if (wave == 0) {  // wave-level TDM issue + completion wait
      if (it == 0) tdm_load_w_tile(&ldsW[0][0], Wtile);
      if (it + 1 < NIT) {
        tdm_load_w_tile(&ldsW[(it + 1) & 1][0], Wtile + k0 + 32);
        __builtin_amdgcn_s_wait_tensorcnt(1);  // current tile complete
      } else {
        __builtin_amdgcn_s_wait_tensorcnt(0);
      }
    }
    __syncthreads();
    const unsigned short* lw = ldsW[it & 1];

    // A fragment: row aRow, 16 contiguous K at k0+lhalf*16
    Frag a;
    const unsigned short* ar = xb + (size_t)aRow * E_ + k0 + lhalf * 16;
    a.q[0] = *(const uint4*)ar;
    a.q[1] = *(const uint4*)(ar + 8);
    __builtin_prefetch(ar + 64, 0, 0);  // A bytes two k-steps ahead

    // B fragments from LDS: lane = column n (l15), 16 contiguous K at lhalf*16
#pragma unroll
    for (int nt = 0; nt < 4; ++nt) {
      Frag bf;
      const unsigned short* bp = &lw[(nt * 16 + l15) * 32 + lhalf * 16];
      bf.q[0] = *(const uint4*)bp;
      bf.q[1] = *(const uint4*)(bp + 8);
      acc[nt] = wmma_bf16(a.v, bf.v, acc[nt]);
    }
    __syncthreads();  // buffer (it&1) free for reload at iteration it+2
  }

  unsigned short* dst = (z == 0) ? Qb : ((z == 1) ? Kb : Vb);
#pragma unroll
  for (int nt = 0; nt < 4; ++nt) {
#pragma unroll
    for (int vr = 0; vr < 8; ++vr) {
      const int m = vr + 8 * lhalf;  // C layout: row
      const int r = m0 + wave * 16 + m;
      const int b = r >> 11, l = r & (L_ - 1);
      const int j = n0 + nt * 16 + l15;  // C layout: col
      const int h = j >> 6, d = j & 63;
      const unsigned short val = f2bf(acc[nt][vr] * oscale);
      size_t idx;
      if (z == 2) idx = (((size_t)b * H_ + h) * D_ + d) * L_ + l;  // V^T
      else        idx = (((size_t)b * H_ + h) * L_ + l) * D_ + d;
      dst[idx] = val;
    }
  }
}

// ---------------------------------------------------------------------------
// Kernel 2: causal flash attention. Per wave: 16 query rows, K-loop step 32.
// Block: 128 threads (4 waves) -> 64 query rows. Grid: (B*H, L/64).
// ---------------------------------------------------------------------------
__global__ __launch_bounds__(128) void flash_attn_kernel(
    const unsigned short* __restrict__ Qb, const unsigned short* __restrict__ Kb,
    const unsigned short* __restrict__ Vt, unsigned short* __restrict__ Ob) {
  __shared__ __align__(16) unsigned short pbuf[4][16 * 32];
  const int bh = blockIdx.x;
  const int wave = threadIdx.x >> 5;
  const int lane = threadIdx.x & 31;
  const int lhalf = lane >> 4, l15 = lane & 15;
  const int qrow0 = blockIdx.y * 64 + wave * 16;
  const unsigned short* Qp = Qb + (size_t)bh * L_ * D_;
  const unsigned short* Kp = Kb + (size_t)bh * L_ * D_;
  const unsigned short* Vp = Vt + (size_t)bh * D_ * L_;
  unsigned short* pb = pbuf[wave];

  // Q A-frags over d (two K=32 halves), loaded once
  Frag aq0, aq1;
  const unsigned short* qptr = Qp + (size_t)(qrow0 + l15) * D_ + lhalf * 16;
  aq0.q[0] = *(const uint4*)qptr;
  aq0.q[1] = *(const uint4*)(qptr + 8);
  aq1.q[0] = *(const uint4*)(qptr + 32);
  aq1.q[1] = *(const uint4*)(qptr + 40);

  v8f o[4] = {};
  float rmax[8], rsum[8];
#pragma unroll
  for (int vr = 0; vr < 8; ++vr) { rmax[vr] = -1e30f; rsum[vr] = 0.0f; }

  const int qhi = qrow0 + 15;
  for (int k0 = 0; k0 <= qhi; k0 += 32) {
    // K^T B-frags: column = key index, rows = d (contiguous in (B,H,L,D))
    Frag bk0a, bk0b, bk1a, bk1b;
    const unsigned short* kp0 = Kp + (size_t)(k0 + l15) * D_ + lhalf * 16;
    bk0a.q[0] = *(const uint4*)kp0;        bk0a.q[1] = *(const uint4*)(kp0 + 8);
    bk0b.q[0] = *(const uint4*)(kp0 + 32); bk0b.q[1] = *(const uint4*)(kp0 + 40);
    const unsigned short* kp1 = kp0 + 16 * D_;
    bk1a.q[0] = *(const uint4*)kp1;        bk1a.q[1] = *(const uint4*)(kp1 + 8);
    bk1b.q[0] = *(const uint4*)(kp1 + 32); bk1b.q[1] = *(const uint4*)(kp1 + 40);

    v8f s0 = {}, s1 = {};
    s0 = wmma_bf16(aq0.v, bk0a.v, s0);
    s0 = wmma_bf16(aq1.v, bk0b.v, s0);
    s1 = wmma_bf16(aq0.v, bk1a.v, s1);
    s1 = wmma_bf16(aq1.v, bk1b.v, s1);

    // causal mask + online softmax (row state per C-row VGPR, replicated
    // across the 16-lane half via xor reductions)
#pragma unroll
    for (int vr = 0; vr < 8; ++vr) {
      const int qr = qrow0 + vr + 8 * lhalf;
      float v0s = ((k0 + l15) <= qr)      ? s0[vr] : -1e30f;
      float v1s = ((k0 + 16 + l15) <= qr) ? s1[vr] : -1e30f;
      float tm = fmaxf(v0s, v1s);
      tm = fmaxf(tm, __shfl_xor(tm, 1));
      tm = fmaxf(tm, __shfl_xor(tm, 2));
      tm = fmaxf(tm, __shfl_xor(tm, 4));
      tm = fmaxf(tm, __shfl_xor(tm, 8));
      const float nm = fmaxf(rmax[vr], tm);
      const float alpha = __expf(rmax[vr] - nm);
      const float p0 = __expf(v0s - nm);
      const float p1 = __expf(v1s - nm);
      float ps = p0 + p1;
      ps += __shfl_xor(ps, 1); ps += __shfl_xor(ps, 2);
      ps += __shfl_xor(ps, 4); ps += __shfl_xor(ps, 8);
      rsum[vr] = rsum[vr] * alpha + ps;
      rmax[vr] = nm;
      o[0][vr] *= alpha; o[1][vr] *= alpha;
      o[2][vr] *= alpha; o[3][vr] *= alpha;
      const int m = vr + 8 * lhalf;
      pb[m * 32 + l15] = f2bf(p0);
      pb[m * 32 + 16 + l15] = f2bf(p1);
    }
    asm volatile("s_wait_dscnt 0" ::: "memory");  // intra-wave LDS hand-off

    // P: C-layout -> A-layout via LDS round trip
    Frag pa;
    const unsigned short* pr = pb + l15 * 32 + lhalf * 16;
    pa.q[0] = *(const uint4*)pr;
    pa.q[1] = *(const uint4*)(pr + 8);

    // O += P @ V  (V stored transposed: B column d is contiguous over k)
#pragma unroll
    for (int dt = 0; dt < 4; ++dt) {
      Frag bv;
      const unsigned short* vp =
          Vp + (size_t)(dt * 16 + l15) * L_ + k0 + lhalf * 16;
      bv.q[0] = *(const uint4*)vp;
      bv.q[1] = *(const uint4*)(vp + 8);
      o[dt] = wmma_bf16(pa.v, bv.v, o[dt]);
    }
  }

#pragma unroll
  for (int vr = 0; vr < 8; ++vr) {
    const int qr = qrow0 + vr + 8 * lhalf;
    const float inv = 1.0f / rsum[vr];
#pragma unroll
    for (int dt = 0; dt < 4; ++dt)
      Ob[((size_t)bh * L_ + qr) * D_ + dt * 16 + l15] = f2bf(o[dt][vr] * inv);
  }
}

// ---------------------------------------------------------------------------
// Kernel 3: output projection. out = O @ Wo^T + bo (fp32 result).
// O is bf16 in (B,H,L,D); K octets are 8-aligned so they never cross a
// head's 64-wide block -> contiguous loads. Wo tile staged via TDM.
// ---------------------------------------------------------------------------
__global__ __launch_bounds__(256) void out_proj_kernel(
    const unsigned short* __restrict__ Ob, const unsigned short* __restrict__ Wob,
    const float* __restrict__ bo, float* __restrict__ out) {
  __shared__ __align__(16) unsigned short ldsW[2][64 * 32];
  const int m0 = blockIdx.x * 128;
  const int n0 = blockIdx.y * 64;
  const int tid = threadIdx.x;
  const int wave = tid >> 5, lane = tid & 31;
  const int lhalf = lane >> 4, l15 = lane & 15;
  const int r = m0 + wave * 16 + l15;
  const int b = r >> 11, l = r & (L_ - 1);
  const unsigned short* Wtile = Wob + (size_t)n0 * E_;

  v8f acc[4] = {};
  const int NIT = E_ / 32;
  for (int it = 0; it < NIT; ++it) {
    const int k0 = it * 32;
    if (wave == 0) {
      if (it == 0) tdm_load_w_tile(&ldsW[0][0], Wtile);
      if (it + 1 < NIT) {
        tdm_load_w_tile(&ldsW[(it + 1) & 1][0], Wtile + k0 + 32);
        __builtin_amdgcn_s_wait_tensorcnt(1);
      } else {
        __builtin_amdgcn_s_wait_tensorcnt(0);
      }
    }
    __syncthreads();
    const unsigned short* lw = ldsW[it & 1];

    Frag a;
    {
      const int kg0 = k0 + lhalf * 16;
      const int kg1 = kg0 + 8;
      const unsigned short* p0 =
          Ob + (((size_t)b * H_ + (kg0 >> 6)) * L_ + l) * D_ + (kg0 & 63);
      const unsigned short* p1 =
          Ob + (((size_t)b * H_ + (kg1 >> 6)) * L_ + l) * D_ + (kg1 & 63);
      a.q[0] = *(const uint4*)p0;
      a.q[1] = *(const uint4*)p1;
      __builtin_prefetch(p0 + 64, 0, 0);
    }

#pragma unroll
    for (int nt = 0; nt < 4; ++nt) {
      Frag bf;
      const unsigned short* bp = &lw[(nt * 16 + l15) * 32 + lhalf * 16];
      bf.q[0] = *(const uint4*)bp;
      bf.q[1] = *(const uint4*)(bp + 8);
      acc[nt] = wmma_bf16(a.v, bf.v, acc[nt]);
    }
    __syncthreads();
  }

#pragma unroll
  for (int nt = 0; nt < 4; ++nt) {
    const int j = n0 + nt * 16 + l15;
    const float bias = bo[j];
#pragma unroll
    for (int vr = 0; vr < 8; ++vr) {
      const int rr = m0 + wave * 16 + vr + 8 * lhalf;
      out[(size_t)rr * E_ + j] = acc[nt][vr] + bias;
    }
  }
}

// ---------------------------------------------------------------------------
extern "C" void kernel_launch(void* const* d_in, const int* in_sizes, int n_in,
                              void* d_out, int out_size, void* d_ws, size_t ws_size,
                              hipStream_t stream) {
  (void)in_sizes; (void)n_in; (void)out_size; (void)ws_size;
  const float* x  = (const float*)d_in[0];
  const float* am = (const float*)d_in[1];
  const float* Wq = (const float*)d_in[2];
  const float* Wk = (const float*)d_in[3];
  const float* Wv = (const float*)d_in[4];
  const float* Wo = (const float*)d_in[5];
  const float* bo = (const float*)d_in[6];
  float* out = (float*)d_out;

  const size_t NE = (size_t)B_ * H_ * L_ * D_;  // 4,194,304 elems
  const size_t WN = (size_t)E_ * E_;            // 1,048,576 elems
  unsigned short* Qb  = (unsigned short*)d_ws;
  unsigned short* Kb  = Qb + NE;
  unsigned short* Vt  = Kb + NE;   // transposed (B,H,D,L)
  unsigned short* Obf = Vt + NE;
  unsigned short* xb  = Obf + NE;  // bf16 x * mask
  unsigned short* Wb  = xb + NE;   // 4 x (E*E) bf16: Wq,Wk,Wv,Wo

  cvt_w_kernel<<<dim3(WN / (256 * 8), 4), 256, 0, stream>>>(Wq, Wk, Wv, Wo, Wb);
  cvt_x_kernel<<<dim3(NE / (256 * 8)), 256, 0, stream>>>(x, am, xb);
  qkv_proj_kernel<<<dim3(B_ * L_ / 128, E_ / 64, 3), 256, 0, stream>>>(
      xb, Wb, Wb + WN, Wb + 2 * WN, Qb, Kb, Vt);
  flash_attn_kernel<<<dim3(B_ * H_, L_ / 64), 128, 0, stream>>>(Qb, Kb, Vt, Obf);
  out_proj_kernel<<<dim3(B_ * L_ / 128, E_ / 64), 256, 0, stream>>>(
      Obf, Wb + 3 * WN, bo, out);
}
